// MinGRU_29703993819750
// MI455X (gfx1250) — compile-verified, matrix operand order checked by reading, not run on hardware
//
#include <hip/hip_runtime.h>
#include <stdint.h>

#define B_   8
#define S_   4096
#define DX_  1024
#define DH_  1024
#define M_   (B_ * S_)        // 32768 rows
#define CHUNK 512
#define NCHUNK (S_ / CHUNK)   // 8

typedef __attribute__((ext_vector_type(16))) __bf16          v16bf;
typedef __attribute__((ext_vector_type(16))) unsigned short  v16u;
typedef __attribute__((ext_vector_type(8)))  float           v8f;

union FragU { uint4 q[2]; v16u u16; v16bf bf; };

__device__ __forceinline__ unsigned short f2bf(float f) {
  union { float f; uint32_t u; } c; c.f = f;
  uint32_t r = c.u + 0x7FFFu + ((c.u >> 16) & 1u);   // RNE
  return (unsigned short)(r >> 16);
}

__device__ __forceinline__ float softplus_f(float x) {
  return fmaxf(x, 0.0f) + log1pf(__expf(-fabsf(x)));   // stable softplus
}

__device__ __forceinline__ float lae(float x, float y) {   // logaddexp
  float mx = fmaxf(x, y);
  float mn = fminf(x, y);
  if (mx == -INFINITY) return -INFINITY;
  return mx + log1pf(__expf(mn - mx));
}

// ---------------------------------------------------------------------------
// CDNA5 async global -> LDS DMA (ASYNCcnt), 16 bytes per lane per op.
// LDS operand = low 32 bits of the generic shared pointer (aperture mapping:
// generic shared addr[31:0] is the LDS offset).
// ---------------------------------------------------------------------------
__device__ __forceinline__ void async_b128(const void* g, void* l) {
  unsigned long long ga = (unsigned long long)g;
  unsigned           lo = (unsigned)(unsigned long long)l;
  asm volatile("global_load_async_to_lds_b128 %0, %1, off"
               :: "v"(lo), "v"(ga) : "memory");
}
#define WAIT_ASYNC_0() asm volatile("s_wait_asynccnt 0x0" ::: "memory")
#define WAIT_ASYNC_4() asm volatile("s_wait_asynccnt 0x4" ::: "memory")

// ---------------------------------------------------------------------------
// f32 -> bf16 conversion, 4 elements / thread
// ---------------------------------------------------------------------------
__global__ void cvt_bf16x4(const float4* __restrict__ s, uint2* __restrict__ d, int n4) {
  int i = blockIdx.x * blockDim.x + threadIdx.x;
  if (i < n4) {
    float4 v = s[i];
    uint2 o;
    o.x = (uint32_t)f2bf(v.x) | ((uint32_t)f2bf(v.y) << 16);
    o.y = (uint32_t)f2bf(v.z) | ((uint32_t)f2bf(v.w) << 16);
    d[i] = o;
  }
}

__device__ __forceinline__ v16bf load_frag(const unsigned short* __restrict__ p) {
  FragU f;
  f.q[0] = *(const uint4*)(p);
  f.q[1] = *(const uint4*)(p + 16);
  return f.bf;
}

// ---------------------------------------------------------------------------
// Fused gate GEMMs: k = x@Wz^T + bz, th = x@Wh^T + bh, log-space epilogue:
//   log_a = -softplus(k), log_v = -softplus(-k) + log_g(th)
// Block tile 128(M) x 64(N), BK=32, 8 waves as 4(M) x 2(N), wave tile 32x32.
// Double-buffered LDS fed by async global->LDS DMA.
// ---------------------------------------------------------------------------
__launch_bounds__(256)
__global__ void proj_gate_kernel(const unsigned short* __restrict__ xg,
                                 const unsigned short* __restrict__ wzg,
                                 const unsigned short* __restrict__ whg,
                                 const float* __restrict__ bz,
                                 const float* __restrict__ bh,
                                 float* __restrict__ log_a,
                                 float* __restrict__ log_v) {
  __shared__ unsigned short xs[2][128 * 40];   // 80B row stride (16B aligned)
  __shared__ unsigned short zs[2][64 * 40];
  __shared__ unsigned short hs[2][64 * 40];

  const int tid  = threadIdx.x;
  const int lane = tid & 31;
  const int w    = tid >> 5;
  const int wm   = w & 3;       // wave M index 0..3
  const int wn   = w >> 2;      // wave N index 0..1
  const int m0   = blockIdx.y * 128;
  const int n0   = blockIdx.x * 64;

  const v8f vzero = {0.f, 0.f, 0.f, 0.f, 0.f, 0.f, 0.f, 0.f};
  v8f accz[2][2], acch[2][2];
#pragma unroll
  for (int i = 0; i < 2; ++i)
#pragma unroll
    for (int j = 0; j < 2; ++j) { accz[i][j] = vzero; acch[i][j] = vzero; }

  const int fr = lane & 15;           // fragment row (M for A, N for B)
  const int kb = (lane >> 4) * 8;     // K sub-block per ISA 16-bit layout

  // per-thread DMA assignments (4 x 16B ops per K-tile)
  const int wrow = tid >> 2, wseg = tid & 3;     // weights: 64 rows x 4 chunks
  auto issue_tile = [&](int k0, int buf) {
#pragma unroll
    for (int t = 0; t < 2; ++t) {                // x tile: 128 rows x 4 chunks
      int c = tid + t * 256;
      int r = c >> 2, s = c & 3;
      async_b128(xg + (size_t)(m0 + r) * DX_ + k0 + s * 8,
                 &xs[buf][r * 40 + s * 8]);
    }
    async_b128(wzg + (size_t)(n0 + wrow) * DX_ + k0 + wseg * 8,
               &zs[buf][wrow * 40 + wseg * 8]);
    async_b128(whg + (size_t)(n0 + wrow) * DX_ + k0 + wseg * 8,
               &hs[buf][wrow * 40 + wseg * 8]);
  };

  issue_tile(0, 0);                              // prologue: fill buffer 0

  for (int k0 = 0; k0 < DX_; k0 += 32) {
    const int cur = (k0 >> 5) & 1;
    const bool has_next = (k0 + 32) < DX_;
    if (has_next) {
      issue_tile(k0 + 32, cur ^ 1);              // DMA next tile into other buf
      WAIT_ASYNC_4();                            // current tile's 4 ops retired
    } else {
      WAIT_ASYNC_0();
    }
    __syncthreads();                             // whole tile valid for all waves

    v16bf A[2], Bz[2], Bh[2];
#pragma unroll
    for (int i = 0; i < 2; ++i)
      A[i] = load_frag(&xs[cur][(wm * 32 + i * 16 + fr) * 40 + kb]);
#pragma unroll
    for (int j = 0; j < 2; ++j) {
      Bz[j] = load_frag(&zs[cur][(wn * 32 + j * 16 + fr) * 40 + kb]);
      Bh[j] = load_frag(&hs[cur][(wn * 32 + j * 16 + fr) * 40 + kb]);
    }
#pragma unroll
    for (int i = 0; i < 2; ++i)
#pragma unroll
      for (int j = 0; j < 2; ++j) {
        accz[i][j] = __builtin_amdgcn_wmma_f32_16x16x32_bf16(
            false, A[i], false, Bz[j], (short)0, accz[i][j], false, false);
        acch[i][j] = __builtin_amdgcn_wmma_f32_16x16x32_bf16(
            false, A[i], false, Bh[j], (short)0, acch[i][j], false, false);
      }
    __syncthreads();                             // done reading buf before reuse
  }

  const int mrow = (lane >> 4) * 8;   // C/D layout: lanes 16-31 are M+8
  const int ncol = lane & 15;
#pragma unroll
  for (int i = 0; i < 2; ++i)
#pragma unroll
    for (int j = 0; j < 2; ++j) {
      int nn = n0 + wn * 32 + j * 16 + ncol;
      float bzv = bz[nn], bhv = bh[nn];
#pragma unroll
      for (int r = 0; r < 8; ++r) {
        int mm = m0 + wm * 32 + i * 16 + mrow + r;
        float kk = accz[i][j][r] + bzv;
        float th = acch[i][j][r] + bhv;
        float la = -softplus_f(kk);                            // log(1-z)
        float lz = -softplus_f(-kk);                           // log z
        float lg = (th >= 0.0f) ? __logf(th + 0.5f) : -softplus_f(-th);
        size_t off = (size_t)mm * DH_ + nn;
        log_a[off] = la;
        log_v[off] = lz + lg;
      }
    }
}

// ---------------------------------------------------------------------------
// Chunked log-space scan: h_t = exp(logaddexp(Lh_{t-1} + log_a_t, log_v_t))
// ---------------------------------------------------------------------------
__global__ void scan_pass1(const float* __restrict__ log_a,
                           const float* __restrict__ log_v,
                           float* __restrict__ cA, float* __restrict__ cV) {
  int idx = blockIdx.x * blockDim.x + threadIdx.x;   // (b,c,h): h fastest
  int h = idx & (DH_ - 1);
  int rest = idx >> 10;
  int c = rest & (NCHUNK - 1);
  int b = rest >> 3;
  size_t base = ((size_t)b * S_ + (size_t)c * CHUNK) * DH_ + h;
  float A = 0.0f, V = -INFINITY;
  for (int s = 0; s < CHUNK; ++s) {
    float la = log_a[base];
    float lv = log_v[base];
    A += la;
    V = lae(V + la, lv);
    base += DH_;
  }
  cA[idx] = A;
  cV[idx] = V;
}

__global__ void scan_pass2(const float* __restrict__ cA,
                           const float* __restrict__ cV,
                           float* __restrict__ cP) {
  int idx = blockIdx.x * blockDim.x + threadIdx.x;   // 8192
  int h = idx & (DH_ - 1);
  int b = idx >> 10;
  float Lh = -INFINITY;
  for (int c = 0; c < NCHUNK; ++c) {
    int o = (b * NCHUNK + c) * DH_ + h;
    cP[o] = Lh;                      // exclusive prefix carry into this chunk
    Lh = lae(Lh + cA[o], cV[o]);
  }
}

__global__ void scan_pass3(const float* __restrict__ log_a,
                           const float* __restrict__ log_v,
                           const float* __restrict__ cP,
                           unsigned short* __restrict__ hout) {
  int idx = blockIdx.x * blockDim.x + threadIdx.x;
  int h = idx & (DH_ - 1);
  int rest = idx >> 10;
  int c = rest & (NCHUNK - 1);
  int b = rest >> 3;
  size_t base = ((size_t)b * S_ + (size_t)c * CHUNK) * DH_ + h;
  float Lh = cP[idx];
  for (int s = 0; s < CHUNK; ++s) {
    float la = log_a[base];
    float lv = log_v[base];
    Lh = lae(Lh + la, lv);
    hout[base] = f2bf(__expf(Lh));
    base += DH_;
  }
}

// ---------------------------------------------------------------------------
// Output GEMM: out = h @ Wo^T + bo.  Block 128x128, BK=32, wave tile 32x64.
// Double-buffered LDS fed by async global->LDS DMA.
// ---------------------------------------------------------------------------
__launch_bounds__(256)
__global__ void out_gemm_kernel(const unsigned short* __restrict__ hg,
                                const unsigned short* __restrict__ wog,
                                const float* __restrict__ bo,
                                float* __restrict__ out) {
  __shared__ unsigned short hsm[2][128 * 40];
  __shared__ unsigned short osm[2][128 * 40];

  const int tid  = threadIdx.x;
  const int lane = tid & 31;
  const int w    = tid >> 5;
  const int wm   = w & 3;       // 0..3
  const int wn   = w >> 2;      // 0..1, wave tile N = 64
  const int m0   = blockIdx.y * 128;
  const int n0   = blockIdx.x * 128;

  const v8f vzero = {0.f, 0.f, 0.f, 0.f, 0.f, 0.f, 0.f, 0.f};
  v8f acc[2][4];
#pragma unroll
  for (int i = 0; i < 2; ++i)
#pragma unroll
    for (int j = 0; j < 4; ++j) acc[i][j] = vzero;

  const int fr = lane & 15;
  const int kb = (lane >> 4) * 8;

  auto issue_tile = [&](int k0, int buf) {
#pragma unroll
    for (int t = 0; t < 2; ++t) {
      int c = tid + t * 256;
      int r = c >> 2, s = c & 3;
      async_b128(hg + (size_t)(m0 + r) * DH_ + k0 + s * 8,
                 &hsm[buf][r * 40 + s * 8]);
      async_b128(wog + (size_t)(n0 + r) * DH_ + k0 + s * 8,
                 &osm[buf][r * 40 + s * 8]);
    }
  };

  issue_tile(0, 0);

  for (int k0 = 0; k0 < DH_; k0 += 32) {
    const int cur = (k0 >> 5) & 1;
    const bool has_next = (k0 + 32) < DH_;
    if (has_next) {
      issue_tile(k0 + 32, cur ^ 1);
      WAIT_ASYNC_4();
    } else {
      WAIT_ASYNC_0();
    }
    __syncthreads();

    v16bf A[2], Bw[4];
#pragma unroll
    for (int i = 0; i < 2; ++i)
      A[i] = load_frag(&hsm[cur][(wm * 32 + i * 16 + fr) * 40 + kb]);
#pragma unroll
    for (int j = 0; j < 4; ++j)
      Bw[j] = load_frag(&osm[cur][(wn * 64 + j * 16 + fr) * 40 + kb]);
#pragma unroll
    for (int i = 0; i < 2; ++i)
#pragma unroll
      for (int j = 0; j < 4; ++j)
        acc[i][j] = __builtin_amdgcn_wmma_f32_16x16x32_bf16(
            false, A[i], false, Bw[j], (short)0, acc[i][j], false, false);
    __syncthreads();
  }

  const int mrow = (lane >> 4) * 8;
  const int ncol = lane & 15;
#pragma unroll
  for (int i = 0; i < 2; ++i)
#pragma unroll
    for (int j = 0; j < 4; ++j) {
      int nn = n0 + wn * 64 + j * 16 + ncol;
      float bov = bo[nn];
#pragma unroll
      for (int r = 0; r < 8; ++r) {
        int mm = m0 + wm * 32 + i * 16 + mrow + r;
        out[(size_t)mm * DX_ + nn] = acc[i][j][r] + bov;
      }
    }
}

// ---------------------------------------------------------------------------
extern "C" void kernel_launch(void* const* d_in, const int* in_sizes, int n_in,
                              void* d_out, int out_size, void* d_ws, size_t ws_size,
                              hipStream_t stream) {
  (void)in_sizes; (void)n_in; (void)out_size; (void)ws_size;
  const float* x  = (const float*)d_in[0];
  const float* Wz = (const float*)d_in[1];
  const float* bz = (const float*)d_in[2];
  const float* Wh = (const float*)d_in[3];
  const float* bh = (const float*)d_in[4];
  const float* Wo = (const float*)d_in[5];
  const float* bo = (const float*)d_in[6];
  float* out = (float*)d_out;

  char* base = (char*)d_ws;
  size_t off = 0;
  auto alloc = [&](size_t bytes) -> char* {
    char* p = base + off;
    off += (bytes + 255) & ~(size_t)255;
    return p;
  };
  unsigned short* xb  = (unsigned short*)alloc((size_t)M_ * DX_ * 2);   //  64 MB
  unsigned short* wzb = (unsigned short*)alloc((size_t)DH_ * DX_ * 2);  //   2 MB
  unsigned short* whb = (unsigned short*)alloc((size_t)DH_ * DX_ * 2);  //   2 MB
  unsigned short* wob = (unsigned short*)alloc((size_t)DX_ * DH_ * 2);  //   2 MB
  float* log_a        = (float*)alloc((size_t)M_ * DH_ * 4);            // 128 MB
  float* log_v        = (float*)alloc((size_t)M_ * DH_ * 4);            // 128 MB
  unsigned short* hb  = (unsigned short*)alloc((size_t)M_ * DH_ * 2);   //  64 MB
  float* cA           = (float*)alloc((size_t)B_ * NCHUNK * DH_ * 4);   // 256 KB
  float* cV           = (float*)alloc((size_t)B_ * NCHUNK * DH_ * 4);
  float* cP           = (float*)alloc((size_t)B_ * NCHUNK * DH_ * 4);

  // 1) f32 -> bf16 conversions
  {
    int n4 = M_ * DX_ / 4;
    cvt_bf16x4<<<(n4 + 255) / 256, 256, 0, stream>>>((const float4*)x, (uint2*)xb, n4);
    int w4 = DH_ * DX_ / 4;
    cvt_bf16x4<<<(w4 + 255) / 256, 256, 0, stream>>>((const float4*)Wz, (uint2*)wzb, w4);
    cvt_bf16x4<<<(w4 + 255) / 256, 256, 0, stream>>>((const float4*)Wh, (uint2*)whb, w4);
    cvt_bf16x4<<<(w4 + 255) / 256, 256, 0, stream>>>((const float4*)Wo, (uint2*)wob, w4);
  }

  // 2) fused gate projections + log-space epilogue
  proj_gate_kernel<<<dim3(DH_ / 64, M_ / 128), 256, 0, stream>>>(
      xb, wzb, whb, bz, bh, log_a, log_v);

  // 3) chunked log-space associative scan over S
  scan_pass1<<<(B_ * NCHUNK * DH_) / 256, 256, 0, stream>>>(log_a, log_v, cA, cV);
  scan_pass2<<<(B_ * DH_) / 256, 256, 0, stream>>>(cA, cV, cP);
  scan_pass3<<<(B_ * NCHUNK * DH_) / 256, 256, 0, stream>>>(log_a, log_v, cP, hb);

  // 4) output projection
  out_gemm_kernel<<<dim3(DX_ / 128, M_ / 128), 256, 0, stream>>>(hb, wob, bo, out);
}
